// SelfAttention_49873160241152
// MI455X (gfx1250) — compile-verified, matrix operand order checked by reading, not run on hardware
//
#include <hip/hip_runtime.h>
#include <math.h>

typedef unsigned short u16;
typedef __attribute__((ext_vector_type(16))) __bf16 bf16x16;
typedef __attribute__((ext_vector_type(8)))  __bf16 bf16x8;
typedef __attribute__((ext_vector_type(8)))  float  f32x8;

#define B_  16
#define S_  1024
#define D_  2048
#define H4_ 8192

#define LDSROW  40                 // padded LDS row stride in u16 (80 B)
#define TILE_AB (128 * LDSROW)     // one 128x32 matrix tile in LDS (u16 elems)
#define BUFSZ   (2 * TILE_AB)      // A tile + B tile per buffer

// ---------- helpers ----------

__device__ __forceinline__ u16 f2bf(float x) {
  unsigned int u = __builtin_bit_cast(unsigned int, x);
  u += 0x7FFFu + ((u >> 16) & 1u);   // round-to-nearest-even
  return (u16)(u >> 16);
}

// Load one 16x32 bf16 WMMA fragment (A-style; B identical with rows = N).
// Memory row-major, K contiguous, ld = K-stride (elements).
// ISA layout: lane L holds row (L&15); K-chunks [8*(L>>4),+8) and +16.
__device__ __forceinline__ bf16x16 load_frag(const u16* p, int ld, int k) {
  int lane = threadIdx.x & 31;
  const u16* q = p + (size_t)(lane & 15) * (size_t)ld + ((lane >> 4) << 3) + k;
  bf16x8 lo = *reinterpret_cast<const bf16x8*>(q);
  bf16x8 hi = *reinterpret_cast<const bf16x8*>(q + 16);
  bf16x16 f;
#pragma unroll
  for (int i = 0; i < 8; ++i) { f[i] = lo[i]; f[i + 8] = hi[i]; }
  return f;
}

// 16 B async copy global -> LDS (per lane), tracked by ASYNCcnt.
__device__ __forceinline__ void async_copy_b128(unsigned ldsoff, const void* g) {
  unsigned long long ga = (unsigned long long)(size_t)g;
  asm volatile("global_load_async_to_lds_b128 %0, %1, off"
               :: "v"(ldsoff), "v"(ga) : "memory");
}

template <int N>
__device__ __forceinline__ void wait_async() {
#if defined(__has_builtin)
#if __has_builtin(__builtin_amdgcn_s_wait_asynccnt)
  __builtin_amdgcn_s_wait_asynccnt(N);
#else
  asm volatile("s_wait_asynccnt %0" :: "n"(N) : "memory");
#endif
#else
  asm volatile("s_wait_asynccnt %0" :: "n"(N) : "memory");
#endif
}

// Fill one LDS buffer: A tile 128x32 + B tile 128x32 (bf16).
// 256 threads, each lane issues 4 x 16B async copies (4 ASYNCcnt per wave).
__device__ __forceinline__ void fill_tile(u16* buf, const u16* Ag, int lda,
                                          const u16* Bg, int ldb, int kk) {
  int tid = threadIdx.x;
#pragma unroll
  for (int q = 0; q < 2; ++q) {
    int id = tid + q * 256;
    int row = id >> 2, ch = id & 3;            // 4 x 16B chunks per row
    async_copy_b128((unsigned)(size_t)(buf + row * LDSROW + ch * 8),
                    Ag + (size_t)row * lda + kk + ch * 8);
  }
#pragma unroll
  for (int q = 0; q < 2; ++q) {
    int id = tid + q * 256;
    int row = id >> 2, ch = id & 3;
    async_copy_b128((unsigned)(size_t)(buf + TILE_AB + row * LDSROW + ch * 8),
                    Bg + (size_t)row * ldb + kk + ch * 8);
  }
}

// Block-level GEMM: C(128x128) += A[128,K] * B[128,K]^T, double-buffered LDS.
// 8 waves in a 2(M) x 4(N) grid; wave tile = 64M x 32N = 4x2 WMMA tiles.
__device__ __forceinline__ void block_gemm_128(const u16* Ag, int lda,
                                               const u16* Bg, int ldb, int K,
                                               u16* lds, f32x8 acc[4][2]) {
  int wave = threadIdx.x >> 5;
  int wm = wave >> 2, wn = wave & 3;
  fill_tile(lds, Ag, lda, Bg, ldb, 0);
  int nk = K >> 5;
  int buf = 0;
  for (int kb = 0; kb < nk; ++kb) {
    if (kb + 1 < nk) {
      fill_tile(lds + (buf ^ 1) * BUFSZ, Ag, lda, Bg, ldb, (kb + 1) << 5);
      wait_async<4>();   // newest 4 = next buffer; older (current) all done
    } else {
      wait_async<0>();
    }
    __syncthreads();
    const u16* lA = lds + buf * BUFSZ;
    const u16* lB = lA + TILE_AB;
    bf16x16 af[4], bfr[2];
#pragma unroll
    for (int i = 0; i < 4; ++i)
      af[i] = load_frag(lA + (wm * 64 + i * 16) * LDSROW, LDSROW, 0);
#pragma unroll
    for (int j = 0; j < 2; ++j)
      bfr[j] = load_frag(lB + (wn * 32 + j * 16) * LDSROW, LDSROW, 0);
#pragma unroll
    for (int i = 0; i < 4; ++i)
#pragma unroll
      for (int j = 0; j < 2; ++j)
        acc[i][j] = __builtin_amdgcn_wmma_f32_16x16x32_bf16(
            false, af[i], false, bfr[j], (short)0, acc[i][j], false, false);
    __syncthreads();
    buf ^= 1;
  }
}

// Register-pipelined wave GEMM for the tiny-M recurrence step (M=16, N=64/wave).
__device__ __forceinline__ void wave_gemm_16x64_pipe(const u16* A, int lda,
                                                     const u16* Bm, int ldb,
                                                     int K, f32x8 acc[4]) {
  bf16x16 a = load_frag(A, lda, 0);
  bf16x16 b[4];
#pragma unroll
  for (int j = 0; j < 4; ++j) b[j] = load_frag(Bm + (size_t)(j * 16) * ldb, ldb, 0);
  for (int k = 0; k < K; k += 32) {
    bf16x16 an{};
    bf16x16 bn[4] = {};
    if (k + 32 < K) {
      an = load_frag(A, lda, k + 32);
#pragma unroll
      for (int j = 0; j < 4; ++j)
        bn[j] = load_frag(Bm + (size_t)(j * 16) * ldb, ldb, k + 32);
    }
#pragma unroll
    for (int j = 0; j < 4; ++j)
      acc[j] = __builtin_amdgcn_wmma_f32_16x16x32_bf16(
          false, a, false, b[j], (short)0, acc[j], false, false);
    a = an;
#pragma unroll
    for (int j = 0; j < 4; ++j) b[j] = bn[j];
  }
}

// ---------- kernels ----------

__global__ void k_f32_to_bf16(const float* __restrict__ src, u16* __restrict__ dst, size_t n) {
  size_t i = (size_t)blockIdx.x * blockDim.x + threadIdx.x;
  size_t stride = (size_t)gridDim.x * blockDim.x;
  for (; i < n; i += stride) dst[i] = f2bf(src[i]);
}

// z=0: Q=(feats Wq^T+bq)/sqrt(D) bf16 [b,s,d]; z=1: K bf16 [b,s,d];
// z=2: V bf16 TRANSPOSED [b,d,s] (B-operand of the PV GEMM).
__global__ void __launch_bounds__(256) k_qkv(
    const u16* __restrict__ feats,
    const u16* __restrict__ Wq, const u16* __restrict__ Wk, const u16* __restrict__ Wv,
    const float* __restrict__ bq, const float* __restrict__ bk, const float* __restrict__ bv,
    u16* __restrict__ Qb, u16* __restrict__ Kb, u16* __restrict__ Vt) {
  __shared__ u16 lds[2 * BUFSZ];
  int z = blockIdx.z;
  size_t m0 = (size_t)blockIdx.y * 128;
  int n0 = blockIdx.x * 128;
  const u16* W = (z == 0) ? Wq : (z == 1) ? Wk : Wv;
  const float* bias = (z == 0) ? bq : (z == 1) ? bk : bv;
  f32x8 acc[4][2] = {};
  block_gemm_128(feats + m0 * D_, D_, W + (size_t)n0 * D_, D_, D_, lds, acc);
  int lane = threadIdx.x & 31, wave = threadIdx.x >> 5;
  int wm = wave >> 2, wn = wave & 3, col = lane & 15, rh = lane >> 4;
  float scale = (z == 0) ? 0.022097086912079608f : 1.0f;  // 1/sqrt(2048)
#pragma unroll
  for (int i = 0; i < 4; ++i)
#pragma unroll
    for (int j = 0; j < 2; ++j) {
      int n = n0 + wn * 32 + j * 16 + col;
      float bb = bias[n];
#pragma unroll
      for (int v = 0; v < 8; ++v) {
        size_t m = m0 + wm * 64 + i * 16 + v + 8 * rh;
        float val = (acc[i][j][v] + bb) * scale;
        if (z == 0)      Qb[m * D_ + n] = f2bf(val);
        else if (z == 1) Kb[m * D_ + n] = f2bf(val);
        else {
          size_t b = m >> 10, s = m & 1023;
          Vt[(b * D_ + (size_t)n) * S_ + s] = f2bf(val);
        }
      }
    }
}

// scores[b,s,t] = Q[b,s,:] . K[b,t,:] (scale folded into Q)
__global__ void __launch_bounds__(256) k_scores(const u16* __restrict__ Qb,
                                                const u16* __restrict__ Kb,
                                                float* __restrict__ Sc) {
  __shared__ u16 lds[2 * BUFSZ];
  size_t z = blockIdx.z;
  size_t m0 = (size_t)blockIdx.y * 128;
  int n0 = blockIdx.x * 128;
  const u16* A  = Qb + z * (size_t)S_ * D_ + m0 * D_;
  const u16* Bm = Kb + z * (size_t)S_ * D_ + (size_t)n0 * D_;
  f32x8 acc[4][2] = {};
  block_gemm_128(A, D_, Bm, D_, D_, lds, acc);
  int lane = threadIdx.x & 31, wave = threadIdx.x >> 5;
  int wm = wave >> 2, wn = wave & 3, col = lane & 15, rh = lane >> 4;
  float* out = Sc + z * (size_t)S_ * S_;
#pragma unroll
  for (int i = 0; i < 4; ++i)
#pragma unroll
    for (int j = 0; j < 2; ++j) {
      int n = n0 + wn * 32 + j * 16 + col;
#pragma unroll
      for (int v = 0; v < 8; ++v) {
        size_t m = m0 + wm * 64 + i * 16 + v + 8 * rh;
        out[m * S_ + n] = acc[i][j][v];
      }
    }
}

// softmax over last dim (1024), bf16 probabilities
__global__ void k_softmax(const float* __restrict__ Sc, u16* __restrict__ P) {
  __shared__ float red[256];
  size_t row = blockIdx.x;
  const float* x = Sc + row * S_;
  int tid = threadIdx.x;
  float vals[4];
  float lmax = -3.4e38f;
#pragma unroll
  for (int i = 0; i < 4; ++i) { vals[i] = x[tid + i * 256]; lmax = fmaxf(lmax, vals[i]); }
  red[tid] = lmax; __syncthreads();
  for (int off = 128; off > 0; off >>= 1) {
    if (tid < off) red[tid] = fmaxf(red[tid], red[tid + off]);
    __syncthreads();
  }
  float mx = red[0]; __syncthreads();
  float lsum = 0.f;
#pragma unroll
  for (int i = 0; i < 4; ++i) { vals[i] = __expf(vals[i] - mx); lsum += vals[i]; }
  red[tid] = lsum; __syncthreads();
  for (int off = 128; off > 0; off >>= 1) {
    if (tid < off) red[tid] += red[tid + off];
    __syncthreads();
  }
  float inv = 1.0f / red[0];
  u16* p = P + row * S_;
#pragma unroll
  for (int i = 0; i < 4; ++i) p[tid + i * 256] = f2bf(vals[i] * inv);
}

// new_feats[b,s,d] = sum_t P[b,s,t] * Vt[b,d,t]
__global__ void __launch_bounds__(256) k_attnout(const u16* __restrict__ P,
                                                 const u16* __restrict__ Vt,
                                                 u16* __restrict__ NF) {
  __shared__ u16 lds[2 * BUFSZ];
  size_t z = blockIdx.z;
  size_t m0 = (size_t)blockIdx.y * 128;
  int n0 = blockIdx.x * 128;
  const u16* A  = P  + z * (size_t)S_ * S_ + m0 * S_;
  const u16* Bm = Vt + z * (size_t)D_ * S_ + (size_t)n0 * S_;
  f32x8 acc[4][2] = {};
  block_gemm_128(A, S_, Bm, S_, S_, lds, acc);
  int lane = threadIdx.x & 31, wave = threadIdx.x >> 5;
  int wm = wave >> 2, wn = wave & 3, col = lane & 15, rh = lane >> 4;
#pragma unroll
  for (int i = 0; i < 4; ++i)
#pragma unroll
    for (int j = 0; j < 2; ++j) {
      int n = n0 + wn * 32 + j * 16 + col;
#pragma unroll
      for (int v = 0; v < 8; ++v) {
        size_t s = m0 + wm * 64 + i * 16 + v + 8 * rh;
        NF[(z * S_ + s) * D_ + n] = f2bf(acc[i][j][v]);
      }
    }
}

// xg[t,b,g] = new_feats[(b,t),:] . W_ih[g,:] + b_ih[g] + b_hh[g]   (fp32)
__global__ void __launch_bounds__(256) k_xg(const u16* __restrict__ NF,
                                            const u16* __restrict__ Wih,
                                            const float* __restrict__ bih,
                                            const float* __restrict__ bhh,
                                            float* __restrict__ XG) {
  __shared__ u16 lds[2 * BUFSZ];
  size_t m0 = (size_t)blockIdx.y * 128;
  int n0 = blockIdx.x * 128;
  f32x8 acc[4][2] = {};
  block_gemm_128(NF + m0 * D_, D_, Wih + (size_t)n0 * D_, D_, D_, lds, acc);
  int lane = threadIdx.x & 31, wave = threadIdx.x >> 5;
  int wm = wave >> 2, wn = wave & 3, col = lane & 15, rh = lane >> 4;
#pragma unroll
  for (int i = 0; i < 4; ++i)
#pragma unroll
    for (int j = 0; j < 2; ++j) {
      int n = n0 + wn * 32 + j * 16 + col;
      float bb = bih[n] + bhh[n];
#pragma unroll
      for (int v = 0; v < 8; ++v) {
        size_t m = m0 + wm * 64 + i * 16 + v + 8 * rh;
        size_t b = m >> 10, s = m & 1023;
        XG[(s * B_ + b) * (size_t)H4_ + n] = acc[i][j][v] + bb;
      }
    }
}

// gates[b,g] = h[b,:] . W_hh[g,:] + xg[t,b,g]   (M = 16 = batch)
__global__ void k_gates(const u16* __restrict__ hbf, const u16* __restrict__ Whh,
                        const float* __restrict__ XG, float* __restrict__ G, int t) {
  int wave = threadIdx.x >> 5, lane = threadIdx.x & 31;
  int n0 = (blockIdx.x * 8 + wave) * 64;
  f32x8 acc[4] = {};
  wave_gemm_16x64_pipe(hbf, D_, Whh + (size_t)n0 * D_, D_, D_, acc);
  int col = lane & 15, rh = lane >> 4;
  const float* xg = XG + (size_t)t * B_ * H4_;
#pragma unroll
  for (int j = 0; j < 4; ++j) {
    int n = n0 + j * 16 + col;
#pragma unroll
    for (int v = 0; v < 8; ++v) {
      int b = v + 8 * rh;
      G[(size_t)b * H4_ + n] = acc[j][v] + xg[(size_t)b * H4_ + n];
    }
  }
}

// LSTM cell update; writes h (fp32) into d_out[b,t,:] and bf16 h state
__global__ void k_update(const float* __restrict__ G, float* __restrict__ C,
                         u16* __restrict__ hbf, float* __restrict__ out, int t) {
  int gid = blockIdx.x * blockDim.x + threadIdx.x;   // 0 .. 16*2048-1
  int b = gid >> 11, j = gid & 2047;
  const float* g = G + (size_t)b * H4_;
  float ig = g[j], fg = g[2048 + j], gg = g[4096 + j], og = g[6144 + j];
  float si = 1.f / (1.f + __expf(-ig));
  float sf = 1.f / (1.f + __expf(-fg));
  float so = 1.f / (1.f + __expf(-og));
  size_t ci = (size_t)b * D_ + j;
  float c = sf * C[ci] + si * tanhf(gg);
  C[ci] = c;
  float h = so * tanhf(c);
  hbf[ci] = f2bf(h);
  out[((size_t)b * S_ + t) * D_ + j] = h;
}

// ---------- host ----------

extern "C" void kernel_launch(void* const* d_in, const int* in_sizes, int n_in,
                              void* d_out, int out_size, void* d_ws, size_t ws_size,
                              hipStream_t stream) {
  (void)in_sizes; (void)n_in; (void)out_size; (void)ws_size;
  const float* feats = (const float*)d_in[0];
  const float* Wq  = (const float*)d_in[1];
  const float* bq  = (const float*)d_in[2];
  const float* Wk  = (const float*)d_in[3];
  const float* bk  = (const float*)d_in[4];
  const float* Wv  = (const float*)d_in[5];
  const float* bv  = (const float*)d_in[6];
  const float* Wih = (const float*)d_in[7];
  const float* Whh = (const float*)d_in[8];
  const float* bih = (const float*)d_in[9];
  const float* bhh = (const float*)d_in[10];
  float* out = (float*)d_out;

  char* w = (char*)d_ws;
  size_t off = 0;
  auto alloc = [&](size_t bytes) -> char* {
    char* p = w + off;
    off += (bytes + 255) & ~(size_t)255;
    return p;
  };
  u16*   featsB = (u16*)  alloc((size_t)B_ * S_ * D_ * 2);
  u16*   WqB    = (u16*)  alloc((size_t)D_ * D_ * 2);
  u16*   WkB    = (u16*)  alloc((size_t)D_ * D_ * 2);
  u16*   WvB    = (u16*)  alloc((size_t)D_ * D_ * 2);
  u16*   WihB   = (u16*)  alloc((size_t)H4_ * D_ * 2);
  u16*   WhhB   = (u16*)  alloc((size_t)H4_ * D_ * 2);
  u16*   Qb     = (u16*)  alloc((size_t)B_ * S_ * D_ * 2);
  u16*   Kb     = (u16*)  alloc((size_t)B_ * S_ * D_ * 2);
  u16*   VtB    = (u16*)  alloc((size_t)B_ * D_ * S_ * 2);
  float* Sc     = (float*)alloc((size_t)B_ * S_ * S_ * 4);
  u16*   Pb     = (u16*)  alloc((size_t)B_ * S_ * S_ * 2);
  u16*   NFb    = (u16*)  alloc((size_t)B_ * S_ * D_ * 2);
  float* XG     = (float*)alloc((size_t)S_ * B_ * H4_ * 4);
  float* Gbuf   = (float*)alloc((size_t)B_ * H4_ * 4);
  float* Cst    = (float*)alloc((size_t)B_ * D_ * 4);
  u16*   hbf    = (u16*)  alloc((size_t)B_ * D_ * 2);

  dim3 blk(256);

  // fp32 -> bf16 conversions
  k_f32_to_bf16<<<dim3(2048), blk, 0, stream>>>(feats, featsB, (size_t)B_ * S_ * D_);
  k_f32_to_bf16<<<dim3(1024), blk, 0, stream>>>(Wq,  WqB,  (size_t)D_ * D_);
  k_f32_to_bf16<<<dim3(1024), blk, 0, stream>>>(Wk,  WkB,  (size_t)D_ * D_);
  k_f32_to_bf16<<<dim3(1024), blk, 0, stream>>>(Wv,  WvB,  (size_t)D_ * D_);
  k_f32_to_bf16<<<dim3(2048), blk, 0, stream>>>(Wih, WihB, (size_t)H4_ * D_);
  k_f32_to_bf16<<<dim3(2048), blk, 0, stream>>>(Whh, WhhB, (size_t)H4_ * D_);

  // zero LSTM state (every call: harness does not re-init ws)
  hipMemsetAsync(Cst, 0, (size_t)B_ * D_ * 4, stream);
  hipMemsetAsync(hbf, 0, (size_t)B_ * D_ * 2, stream);

  // attention front end (LDS double-buffered async GEMMs)
  k_qkv    <<<dim3(16, 128, 3), blk, 0, stream>>>(featsB, WqB, WkB, WvB, bq, bk, bv, Qb, Kb, VtB);
  k_scores <<<dim3(8, 8, 16),   blk, 0, stream>>>(Qb, Kb, Sc);
  k_softmax<<<dim3(B_ * S_),    blk, 0, stream>>>(Sc, Pb);
  k_attnout<<<dim3(16, 8, 16),  blk, 0, stream>>>(Pb, VtB, NFb);

  // LSTM input projection (time-major layout for the recurrence)
  k_xg<<<dim3(64, 128, 1), blk, 0, stream>>>(NFb, WihB, bih, bhh, XG);

  // sequential recurrence
  for (int t = 0; t < S_; ++t) {
    k_gates <<<dim3(16),  blk, 0, stream>>>(hbf, WhhB, XG, Gbuf, t);
    k_update<<<dim3(128), blk, 0, stream>>>(Gbuf, Cst, hbf, out, t);
  }
}
